// TransformerEncoderLayer_26156350832795
// MI455X (gfx1250) — compile-verified, matrix operand order checked by reading, not run on hardware
//
#include <hip/hip_runtime.h>
#include <hip/hip_bf16.h>

// ---------------------------------------------------------------------------
// Types for CDNA5 WMMA (wave32): v_wmma_f32_16x16x32_bf16
// ---------------------------------------------------------------------------
typedef __attribute__((ext_vector_type(16))) __bf16 v16bf;
typedef __attribute__((ext_vector_type(8)))  __bf16 v8bf;
typedef __attribute__((ext_vector_type(8)))  float  v8f;
typedef int v4i __attribute__((vector_size(16)));

union BF16x16 { v16bf v; v8bf h[2]; };

__device__ __forceinline__ v8f wmma_bf16(const BF16x16& a, const BF16x16& b, v8f c) {
  return __builtin_amdgcn_wmma_f32_16x16x32_bf16(false, a.v, false, b.v,
                                                 (short)0, c, false, false);
}

#define ACT_NONE 0
#define ACT_GELU 1

// Model constants
#define SEQ_    2048
#define DM_     1024
#define DFF_    4096
#define DIN_    2048
#define DST_    16
#define NH_     16
#define DK_     64
#define BTOT_   4096   // BATCH * SEQ

// ---------------------------------------------------------------------------
// Async global -> LDS staging (CDNA5 GLOBAL_LOAD_ASYNC_TO_LDS_B128, ASYNCcnt)
// with sync fallback if the builtin is absent in this toolchain.
// ---------------------------------------------------------------------------
#if __has_builtin(__builtin_amdgcn_global_load_async_to_lds_b128)
#define HAS_ASYNC_LDS 1
#endif

__device__ __forceinline__ void stage16(__bf16* lds, const __bf16* g) {
#ifdef HAS_ASYNC_LDS
  __builtin_amdgcn_global_load_async_to_lds_b128(
      (__attribute__((address_space(1))) v4i*)const_cast<__bf16*>(g),
      (__attribute__((address_space(3))) v4i*)(void*)lds, 0, 0);
#else
  *(v8bf*)lds = *(const v8bf*)g;
#endif
}

// Wait until at most N async ops remain outstanding (per-wave, in-order).
template <int N>
__device__ __forceinline__ void stage_wait() {
#ifdef HAS_ASYNC_LDS
#if __has_builtin(__builtin_amdgcn_s_wait_asynccnt)
  __builtin_amdgcn_s_wait_asynccnt(N);
#else
  asm volatile("s_wait_asynccnt %0" :: "i"(N) : "memory");
#endif
#endif
}

// ---------------------------------------------------------------------------
// fp32 -> bf16 TRANSPOSED convert: out[Npad x K] = in[K x N]^T, zero-padded
// rows N..Npad-1. One-time; fp32 source fits in 192MB L2, so strided reads OK.
// ---------------------------------------------------------------------------
__global__ void k_cvt_t(const float* __restrict__ in, __bf16* __restrict__ out,
                        int K, int N, int total /* Npad*K */) {
  int idx = blockIdx.x * 256 + threadIdx.x;
  if (idx >= total) return;
  int n = idx / K, k = idx - n * K;
  out[idx] = (n < N) ? (__bf16)in[(size_t)k * N + n] : (__bf16)0.f;
}

// ---------------------------------------------------------------------------
// LayerNorm over rows of 1024, bf16 output (feeds WMMA GEMMs only)
// ---------------------------------------------------------------------------
__global__ __launch_bounds__(256) void k_ln(const float* __restrict__ X,
                                            const float* __restrict__ g,
                                            const float* __restrict__ b,
                                            __bf16* __restrict__ out) {
  const int row = blockIdx.x, tid = threadIdx.x;
  const float* xr = X + (size_t)row * DM_;
  __shared__ float red[256];
  float v[4]; float s = 0.f;
#pragma unroll
  for (int i = 0; i < 4; ++i) { v[i] = xr[tid + i * 256]; s += v[i]; }
  red[tid] = s; __syncthreads();
  for (int st = 128; st > 0; st >>= 1) { if (tid < st) red[tid] += red[tid + st]; __syncthreads(); }
  float mean = red[0] * (1.f / DM_);
  __syncthreads();
  float vs = 0.f;
#pragma unroll
  for (int i = 0; i < 4; ++i) { float d = v[i] - mean; vs += d * d; }
  red[tid] = vs; __syncthreads();
  for (int st = 128; st > 0; st >>= 1) { if (tid < st) red[tid] += red[tid + st]; __syncthreads(); }
  float rstd = rsqrtf(red[0] * (1.f / DM_) + 1e-5f);
#pragma unroll
  for (int i = 0; i < 4; ++i) {
    int c = tid + i * 256;
    out[(size_t)row * DM_ + c] = (__bf16)((v[i] - mean) * rstd * g[c] + b[c]);
  }
}

// ---------------------------------------------------------------------------
// WMMA bf16 GEMM: C[M,N] = act(A[M,K] * Bt[N,K]^T + bias) + resid
// Bt is the PRE-TRANSPOSED weight (N x K row-major).
// Block: 128 threads (4 waves), 64x64 tile, K-step 32. Double-buffered async
// staging with the pipeline unrolled x2 so buffer parity is compile-time
// (K/32 is even for every GEMM in this model). Consume gated by
// s_wait_asynccnt<=4 (the 4 newer in-flight ops) + barrier.
// cbfT: 0 -> Cbf row-major [M][N]; 1 -> Cbf transposed [N][M].
// ---------------------------------------------------------------------------
__global__ __launch_bounds__(128) void k_gemm(
    const __bf16* __restrict__ A, const __bf16* __restrict__ Bt,
    const float* __restrict__ bias, const float* __restrict__ resid,
    float* __restrict__ Cf, __bf16* __restrict__ Cbf,
    int M, int N, int K, int act, int cbfT) {
  __shared__ __bf16 As[2][64][32];
  __shared__ __bf16 Bs[2][64][32];   // Bs[buf][n][k]
  const int tid = threadIdx.x;
  const int lane = tid & 31, wave = tid >> 5;
  const int lrow = lane & 15, lhalf = lane >> 4;
  const int wm = wave >> 1, wn = wave & 1;
  const int m0 = blockIdx.y * 64, n0 = blockIdx.x * 64;

  // per-thread staging assignment: 2 fixed 16B chunks per tile side
  const int sr = tid >> 2;            // 0..31
  const int sk = (tid & 3) * 8;       // 0,8,16,24
  const __bf16* gA0 = A  + (size_t)(m0 + sr)      * K + sk;
  const __bf16* gA1 = A  + (size_t)(m0 + sr + 32) * K + sk;
  const __bf16* gB0 = Bt + (size_t)(n0 + sr)      * K + sk;
  const __bf16* gB1 = Bt + (size_t)(n0 + sr + 32) * K + sk;

  v8f acc[2][2];
#pragma unroll
  for (int mt = 0; mt < 2; ++mt)
#pragma unroll
    for (int nt = 0; nt < 2; ++nt)
#pragma unroll
      for (int i = 0; i < 8; ++i) acc[mt][nt][i] = 0.f;

#define GEMM_ISSUE(buf)                                              \
  do {                                                               \
    stage16(&As[buf][sr][sk],      gA0);                             \
    stage16(&As[buf][sr + 32][sk], gA1);                             \
    stage16(&Bs[buf][sr][sk],      gB0);                             \
    stage16(&Bs[buf][sr + 32][sk], gB1);                             \
    gA0 += 32; gA1 += 32; gB0 += 32; gB1 += 32;                      \
  } while (0)

#define GEMM_COMPUTE(buf)                                            \
  do {                                                               \
    BF16x16 af[2], bfr[2];                                           \
    _Pragma("unroll")                                                \
    for (int mt = 0; mt < 2; ++mt) {                                 \
      int r = wm * 32 + mt * 16 + lrow;                              \
      af[mt].h[0] = *(const v8bf*)&As[buf][r][lhalf * 8];            \
      af[mt].h[1] = *(const v8bf*)&As[buf][r][16 + lhalf * 8];       \
    }                                                                \
    _Pragma("unroll")                                                \
    for (int nt = 0; nt < 2; ++nt) {                                 \
      int r = wn * 32 + nt * 16 + lrow;                              \
      bfr[nt].h[0] = *(const v8bf*)&Bs[buf][r][lhalf * 8];           \
      bfr[nt].h[1] = *(const v8bf*)&Bs[buf][r][16 + lhalf * 8];      \
    }                                                                \
    _Pragma("unroll")                                                \
    for (int mt = 0; mt < 2; ++mt)                                   \
      _Pragma("unroll")                                              \
      for (int nt = 0; nt < 2; ++nt)                                 \
        acc[mt][nt] = wmma_bf16(af[mt], bfr[nt], acc[mt][nt]);       \
  } while (0)

  GEMM_ISSUE(0);                      // prologue: tile 0 -> buf 0
  for (int k0 = 0; k0 < K; k0 += 64) {
    GEMM_ISSUE(1);                    // odd tile -> buf 1
    stage_wait<4>();                  // buf 0 landed (4 newer outstanding)
    __syncthreads();
    GEMM_COMPUTE(0);
    __syncthreads();                  // buf0 readers done before reuse
    if (k0 + 64 < K) { GEMM_ISSUE(0); stage_wait<4>(); }
    else             { stage_wait<0>(); }
    __syncthreads();
    GEMM_COMPUTE(1);
    __syncthreads();                  // buf1 readers done before reuse
  }
#undef GEMM_ISSUE
#undef GEMM_COMPUTE

  // epilogue: C layout -> N = lane&15, M = r + 8*(lane>>4)
#pragma unroll
  for (int mt = 0; mt < 2; ++mt)
#pragma unroll
    for (int nt = 0; nt < 2; ++nt) {
      int gcol = n0 + wn * 32 + nt * 16 + lrow;
      if (gcol >= N) continue;
      float bval = bias ? bias[gcol] : 0.f;
#pragma unroll
      for (int r = 0; r < 8; ++r) {
        int grow = m0 + wm * 32 + mt * 16 + r + 8 * lhalf;
        if (grow >= M) continue;
        float v = acc[mt][nt][r] + bval;
        if (act == ACT_GELU) {
          float x3 = v * v * v;
          v = 0.5f * v * (1.f + tanhf(0.7978845608f * (v + 0.044715f * x3)));
        }
        size_t o = (size_t)grow * N + gcol;
        if (resid) v += resid[o];
        if (Cf)  Cf[o] = v;
        if (Cbf) Cbf[cbfT ? ((size_t)gcol * M + grow) : o] = (__bf16)v;
      }
    }
}

// ---------------------------------------------------------------------------
// Flash attention (causal), bf16, WMMA. V supplied TRANSPOSED: VT[dim][b*S+s].
// Grid: (S/64, H, B). Block: 128 (4 waves); wave w owns query rows 16w..16w+15.
// V tiles double-buffered (2 async ops/thread -> wait<=2); pipeline unrolled
// x2 (key-block count nkb = 2*(qblk+1) is always even).
// ---------------------------------------------------------------------------
__device__ __forceinline__ float redmax16(float v) {
  v = fmaxf(v, __shfl_xor(v, 1, 32));
  v = fmaxf(v, __shfl_xor(v, 2, 32));
  v = fmaxf(v, __shfl_xor(v, 4, 32));
  v = fmaxf(v, __shfl_xor(v, 8, 32));
  return v;
}
__device__ __forceinline__ float redsum16(float v) {
  v += __shfl_xor(v, 1, 32);
  v += __shfl_xor(v, 2, 32);
  v += __shfl_xor(v, 4, 32);
  v += __shfl_xor(v, 8, 32);
  return v;
}

__global__ __launch_bounds__(128) void k_attn(
    const __bf16* __restrict__ Q, const __bf16* __restrict__ Kc,
    const __bf16* __restrict__ VT, __bf16* __restrict__ O) {
  const int qblk = blockIdx.x, h = blockIdx.y, b = blockIdx.z;
  const int qbase = qblk * 64;
  const int tid = threadIdx.x;
  const int lane = tid & 31, wave = tid >> 5;
  const int lrow = lane & 15, lhalf = lane >> 4;
  const int ldr = NH_ * DK_;  // 1024
  const size_t base = (size_t)b * SEQ_ * ldr + h * DK_;
  const __bf16* Qp = Q + base;
  const __bf16* Kp = Kc + base;
  __bf16* Op = O + base;

  __shared__ __bf16 Vt[2][64][32];     // [buf][d][key] tiles of VT
  __shared__ __bf16 Pl[4][16][32];     // per-wave P staging (C-layout -> A-layout)

  // V staging: per-thread 2 fixed chunks; VT row = h*64+d, col = b*S + key
  const int sd = tid >> 2;             // 0..31
  const int sk8 = (tid & 3) * 8;       // key chunk
  const __bf16* gV0 = VT + (size_t)(h * 64 + sd)      * BTOT_ + (size_t)b * SEQ_ + sk8;
  const __bf16* gV1 = VT + (size_t)(h * 64 + sd + 32) * BTOT_ + (size_t)b * SEQ_ + sk8;

#define ATTN_ISSUE(buf)                                   \
  do {                                                    \
    stage16(&Vt[buf][sd][sk8],      gV0);                 \
    stage16(&Vt[buf][sd + 32][sk8], gV1);                 \
    gV0 += 32; gV1 += 32;                                 \
  } while (0)

  // Q A-fragments (16 rows x 64 d), prescaled by 1/sqrt(64)
  BF16x16 qa[2];
  {
    int qrow = qbase + wave * 16 + lrow;
#pragma unroll
    for (int g = 0; g < 2; ++g) {
      qa[g].h[0] = *(const v8bf*)(Qp + (size_t)qrow * ldr + g * 32 + lhalf * 8);
      qa[g].h[1] = *(const v8bf*)(Qp + (size_t)qrow * ldr + g * 32 + 16 + lhalf * 8);
#pragma unroll
      for (int i = 0; i < 16; ++i)
        qa[g].v[i] = (__bf16)((float)qa[g].v[i] * 0.125f);
    }
  }

  float m_r[8], l_r[8];
  v8f o_acc[4];
#pragma unroll
  for (int r = 0; r < 8; ++r) { m_r[r] = -1e30f; l_r[r] = 0.f; }
#pragma unroll
  for (int t = 0; t < 4; ++t)
#pragma unroll
    for (int i = 0; i < 8; ++i) o_acc[t][i] = 0.f;

#define ATTN_BODY(kbase_, buf)                                              \
  do {                                                                      \
    v8f st[2];                                                              \
    _Pragma("unroll")                                                       \
    for (int t = 0; t < 2; ++t) {                                           \
      int krow = (kbase_) + t * 16 + lrow;                                  \
      BF16x16 k0, k1;                                                       \
      k0.h[0] = *(const v8bf*)(Kp + (size_t)krow * ldr + lhalf * 8);        \
      k0.h[1] = *(const v8bf*)(Kp + (size_t)krow * ldr + 16 + lhalf * 8);   \
      k1.h[0] = *(const v8bf*)(Kp + (size_t)krow * ldr + 32 + lhalf * 8);   \
      k1.h[1] = *(const v8bf*)(Kp + (size_t)krow * ldr + 48 + lhalf * 8);   \
      v8f s;                                                                \
      _Pragma("unroll")                                                     \
      for (int i = 0; i < 8; ++i) s[i] = 0.f;                               \
      s = wmma_bf16(qa[0], k0, s);                                          \
      s = wmma_bf16(qa[1], k1, s);                                          \
      st[t] = s;                                                            \
    }                                                                       \
    const int key0 = (kbase_) + lrow, key1 = (kbase_) + 16 + lrow;          \
    _Pragma("unroll")                                                       \
    for (int r = 0; r < 8; ++r) {                                           \
      int qr = qbase + wave * 16 + r + 8 * lhalf;                           \
      float s0 = (key0 <= qr) ? st[0][r] : -1e30f;                          \
      float s1 = (key1 <= qr) ? st[1][r] : -1e30f;                          \
      float mx = redmax16(fmaxf(s0, s1));                                   \
      float mnew = fmaxf(m_r[r], mx);                                       \
      float alpha = __expf(m_r[r] - mnew);                                  \
      float p0 = __expf(s0 - mnew);                                         \
      float p1 = __expf(s1 - mnew);                                         \
      l_r[r] = l_r[r] * alpha + redsum16(p0 + p1);                          \
      m_r[r] = mnew;                                                        \
      _Pragma("unroll")                                                     \
      for (int t = 0; t < 4; ++t) o_acc[t][r] = o_acc[t][r] * alpha;        \
      Pl[wave][r + 8 * lhalf][lrow]      = (__bf16)p0;                      \
      Pl[wave][r + 8 * lhalf][16 + lrow] = (__bf16)p1;                      \
    }                                                                       \
    BF16x16 pa;                                                             \
    pa.h[0] = *(const v8bf*)&Pl[wave][lrow][lhalf * 8];                     \
    pa.h[1] = *(const v8bf*)&Pl[wave][lrow][16 + lhalf * 8];                \
    _Pragma("unroll")                                                       \
    for (int t = 0; t < 4; ++t) {                                           \
      BF16x16 vb;                                                           \
      vb.h[0] = *(const v8bf*)&Vt[buf][t * 16 + lrow][lhalf * 8];           \
      vb.h[1] = *(const v8bf*)&Vt[buf][t * 16 + lrow][16 + lhalf * 8];      \
      o_acc[t] = wmma_bf16(pa, vb, o_acc[t]);                               \
    }                                                                       \
  } while (0)

  const int nkb = (qbase + 64) >> 5;   // 32-key blocks; always even
  ATTN_ISSUE(0);                       // prologue: key block 0 -> buf 0
  for (int kb = 0; kb < nkb; kb += 2) {
    ATTN_ISSUE(1);                     // odd block -> buf 1
    stage_wait<2>();                   // buf 0 landed (2 newer outstanding)
    __syncthreads();
    ATTN_BODY(kb * 32, 0);
    __syncthreads();
    if (kb + 2 < nkb) { ATTN_ISSUE(0); stage_wait<2>(); }
    else              { stage_wait<0>(); }
    __syncthreads();
    ATTN_BODY(kb * 32 + 32, 1);
    __syncthreads();
  }
#undef ATTN_ISSUE
#undef ATTN_BODY

  // epilogue: normalize and store bf16 (row-major, feeds Wo GEMM as A)
#pragma unroll
  for (int t = 0; t < 4; ++t) {
    int col = t * 16 + lrow;
#pragma unroll
    for (int r = 0; r < 8; ++r) {
      int qr = qbase + wave * 16 + r + 8 * lhalf;
      Op[(size_t)qr * ldr + col] = (__bf16)(o_acc[t][r] / l_r[r]);
    }
  }
}

// ---------------------------------------------------------------------------
// Depthwise causal conv(4) + bias + SiLU; input xr[:, 0:2048] f32, out bf16
// ---------------------------------------------------------------------------
__global__ void k_conv(const float* __restrict__ xr, const float* __restrict__ w,
                       const float* __restrict__ cb, __bf16* __restrict__ out) {
  int idx = blockIdx.x * 256 + threadIdx.x;            // rows(4096) * 2048
  int c = idx & (DIN_ - 1);
  int row = idx >> 11;
  int s = row & (SEQ_ - 1);
  float acc = cb[c];
#pragma unroll
  for (int t = 0; t < 4; ++t) {
    int sp = s - 3 + t;
    if (sp >= 0) acc += w[c * 4 + t] * xr[(size_t)(row - 3 + t) * (2 * DIN_) + c];
  }
  float sl = acc / (1.f + __expf(-acc));               // SiLU
  out[(size_t)row * DIN_ + c] = (__bf16)sl;
}

// ---------------------------------------------------------------------------
// 16-state sequential scan per batch (1 wave). Precomputes 16 exp(exp(dt)*A).
// ---------------------------------------------------------------------------
__global__ __launch_bounds__(32) void k_scan(const float* __restrict__ xdbl,
                                             const float* __restrict__ dt,
                                             const float* __restrict__ A,
                                             float* __restrict__ hs) {
  const int b = blockIdx.x, t = threadIdx.x;
  __shared__ float Aex[16][16][16];
  __shared__ float hsh[16];
  for (int i = t; i < 4096; i += 32) {
    int m = i >> 8, j = (i >> 4) & 15, k = i & 15;
    Aex[m][j][k] = __expf(__expf(dt[m]) * A[j * 16 + k]);
  }
  if (t < 16) hsh[t] = 0.f;
  __syncthreads();
  for (int s = 0; s < SEQ_; ++s) {
    int m = s & 15;
    float hn = 0.f;
    if (t < 16) {
      float acc = 0.f;
#pragma unroll
      for (int k = 0; k < 16; ++k) acc += hsh[k] * Aex[m][t][k];
      const float* xd = xdbl + ((size_t)b * SEQ_ + s) * (2 * DST_);
      hn = tanhf(acc) + xd[t] * hsh[t] + xd[16 + t];
    }
    __syncthreads();
    if (t < 16) {
      hsh[t] = hn;
      hs[((size_t)b * SEQ_ + s) * DST_ + t] = hn;
    }
    __syncthreads();
  }
}

// ---------------------------------------------------------------------------
// y = repeat(hs,128) + res ; bf16 out (feeds out-proj GEMM)
// ---------------------------------------------------------------------------
__global__ void k_yrep(const float* __restrict__ hs, const float* __restrict__ xr,
                       __bf16* __restrict__ out) {
  int idx = blockIdx.x * 256 + threadIdx.x;            // rows(4096) * 2048
  int c = idx & (DIN_ - 1);
  int row = idx >> 11;
  float y = hs[(size_t)row * DST_ + (c >> 7)] + xr[(size_t)row * (2 * DIN_) + DIN_ + c];
  out[(size_t)row * DIN_ + c] = (__bf16)y;
}

// ---------------------------------------------------------------------------
// Host: orchestration
// ---------------------------------------------------------------------------
extern "C" void kernel_launch(void* const* d_in, const int* in_sizes, int n_in,
                              void* d_out, int out_size, void* d_ws, size_t ws_size,
                              hipStream_t stream) {
  (void)in_sizes; (void)n_in; (void)out_size; (void)ws_size;
  const float* x      = (const float*)d_in[0];
  const float* ln1_g  = (const float*)d_in[1];
  const float* ln1_b  = (const float*)d_in[2];
  const float* Wq     = (const float*)d_in[3];
  const float* bq     = (const float*)d_in[4];
  const float* Wk     = (const float*)d_in[5];
  const float* bk     = (const float*)d_in[6];
  const float* Wv     = (const float*)d_in[7];
  const float* bv     = (const float*)d_in[8];
  const float* Wo     = (const float*)d_in[9];
  const float* bo     = (const float*)d_in[10];
  const float* ln2_g  = (const float*)d_in[11];
  const float* ln2_b  = (const float*)d_in[12];
  const float* in_w   = (const float*)d_in[13];
  const float* in_b   = (const float*)d_in[14];
  const float* conv_w = (const float*)d_in[15];
  const float* conv_b = (const float*)d_in[16];
  const float* xp_w   = (const float*)d_in[17];
  const float* xp_b   = (const float*)d_in[18];
  const float* dt     = (const float*)d_in[19];
  const float* A      = (const float*)d_in[20];
  /* Dp (d_in[21]) unused by the reference math */
  const float* out_w  = (const float*)d_in[22];
  const float* out_b  = (const float*)d_in[23];
  const float* ln3_g  = (const float*)d_in[24];
  const float* ln3_b  = (const float*)d_in[25];
  const float* fc1_w  = (const float*)d_in[26];
  const float* fc1_b  = (const float*)d_in[27];
  const float* fc2_w  = (const float*)d_in[28];
  const float* fc2_b  = (const float*)d_in[29];

  char* ws = (char*)d_ws;
  // bf16 TRANSPOSED weight pool (element offsets); all are [N(pad) x K]
  __bf16* wbf = (__bf16*)ws;
  __bf16* wq_bf   = wbf + 0;         // 1024x1024
  __bf16* wk_bf   = wbf + 1048576;
  __bf16* wv_bf   = wbf + 2097152;
  __bf16* wo_bf   = wbf + 3145728;
  __bf16* inw_bf  = wbf + 4194304;   // 4096x1024
  __bf16* xpw_bf  = wbf + 8388608;   // 64x2048 (rows 32..63 zero-padded)
  __bf16* outw_bf = wbf + 8519680;   // 1024x2048
  __bf16* fc1w_bf = wbf + 10616832;  // 4096x1024
  __bf16* fc2w_bf = wbf + 14811136;  // 1024x4096
  const size_t base1 = 38010880;     // bytes (= 19005440 bf16 elems, 256B aligned)
  __bf16* hbf  = (__bf16*)(ws + base1);                 // 8 MB, reused ln1/ln2/ln3
  float*  x1   = (float*)(ws + base1 + 8388608);        // 16 MB
  float*  x2   = (float*)(ws + base1 + 25165824);       // 16 MB
  float*  xr   = (float*)(ws + base1 + 41943040);       // 64 MB
  char*   rA   = ws + base1 + 109051904;                // 33 MB aliased region
  // phase 1 (attention)
  __bf16* qbf    = (__bf16*)(rA + 0);
  __bf16* kbf    = (__bf16*)(rA + 8388608);
  __bf16* vbfT   = (__bf16*)(rA + 16777216);            // transposed [1024][4096]
  __bf16* attnbf = (__bf16*)(rA + 25165824);
  // phase 2 (SSM branch)
  __bf16* xcbf = (__bf16*)(rA + 0);
  float*  xdbl = (float*)(rA + 16777216);
  float*  hsb  = (float*)(rA + 17301504);
  __bf16* ybf  = (__bf16*)(rA + 17563648);
  // phase 3 (MLP)
  __bf16* gbf  = (__bf16*)(rA + 0);

  auto cvtT = [&](const float* src, __bf16* dst, int K, int N, int Npad) {
    int total = Npad * K;
    k_cvt_t<<<(total + 255) / 256, 256, 0, stream>>>(src, dst, K, N, total);
  };
  cvtT(Wq, wq_bf, DM_, DM_, DM_);
  cvtT(Wk, wk_bf, DM_, DM_, DM_);
  cvtT(Wv, wv_bf, DM_, DM_, DM_);
  cvtT(Wo, wo_bf, DM_, DM_, DM_);
  cvtT(in_w, inw_bf, DM_, 2 * DIN_, 2 * DIN_);
  cvtT(xp_w, xpw_bf, DIN_, 2 * DST_, 64);
  cvtT(out_w, outw_bf, DIN_, DM_, DM_);
  cvtT(fc1_w, fc1w_bf, DM_, DFF_, DFF_);
  cvtT(fc2_w, fc2w_bf, DFF_, DM_, DM_);

  auto gemm = [&](const __bf16* a, const __bf16* bt, const float* bias,
                  const float* resid, float* cf, __bf16* cbf,
                  int M, int N, int K, int act, int cbfT) {
    dim3 grid((N + 63) / 64, (M + 63) / 64);
    k_gemm<<<grid, 128, 0, stream>>>(a, bt, bias, resid, cf, cbf, M, N, K, act, cbfT);
  };

  // ---- attention block ----
  k_ln<<<BTOT_, 256, 0, stream>>>(x, ln1_g, ln1_b, hbf);
  gemm(hbf, wq_bf, bq, nullptr, nullptr, qbf, BTOT_, DM_, DM_, ACT_NONE, 0);
  gemm(hbf, wk_bf, bk, nullptr, nullptr, kbf, BTOT_, DM_, DM_, ACT_NONE, 0);
  gemm(hbf, wv_bf, bv, nullptr, nullptr, vbfT, BTOT_, DM_, DM_, ACT_NONE, 1);
  k_attn<<<dim3(SEQ_ / 64, NH_, 2), 128, 0, stream>>>(qbf, kbf, vbfT, attnbf);
  gemm(attnbf, wo_bf, bo, x, x1, nullptr, BTOT_, DM_, DM_, ACT_NONE, 0);

  // ---- SSM branch ----
  k_ln<<<BTOT_, 256, 0, stream>>>(x1, ln2_g, ln2_b, hbf);
  gemm(hbf, inw_bf, in_b, nullptr, xr, nullptr, BTOT_, 2 * DIN_, DM_, ACT_NONE, 0);
  k_conv<<<(BTOT_ * DIN_) / 256, 256, 0, stream>>>(xr, conv_w, conv_b, xcbf);
  gemm(xcbf, xpw_bf, xp_b, nullptr, xdbl, nullptr, BTOT_, 2 * DST_, DIN_, ACT_NONE, 0);
  k_scan<<<2, 32, 0, stream>>>(xdbl, dt, A, hsb);
  k_yrep<<<(BTOT_ * DIN_) / 256, 256, 0, stream>>>(hsb, xr, ybf);
  gemm(ybf, outw_bf, out_b, x1, x2, nullptr, BTOT_, DM_, DIN_, ACT_NONE, 0);

  // ---- MLP block ----
  k_ln<<<BTOT_, 256, 0, stream>>>(x2, ln3_g, ln3_b, hbf);
  gemm(hbf, fc1w_bf, fc1_b, nullptr, nullptr, gbf, BTOT_, DFF_, DM_, ACT_GELU, 0);
  gemm(gbf, fc2w_bf, fc2_b, x2, (float*)d_out, nullptr, BTOT_, DM_, DFF_, ACT_NONE, 0);
}
